// CausalSelfAttention_5875515261619
// MI455X (gfx1250) — matrix operand order repaired and checked by reading, not compiled, not run on hardware
//
#include <hip/hip_runtime.h>
#include <hip/hip_bf16.h>
#include <stdint.h>

typedef __attribute__((ext_vector_type(16))) _Float16     v16h;
typedef __attribute__((ext_vector_type(8)))  float        v8f;
typedef __attribute__((ext_vector_type(4)))  unsigned int u32x4;
typedef __attribute__((ext_vector_type(4)))  float        f32x4;

union Frag16 { v16h v; _Float16 h[16]; u32x4 q[2]; };

#define Bb 2
#define Tt 2048
#define Cc 1024
#define Hh 16
#define Dd 64
#define Mm (Bb*Tt)

// Async copy of 32 bytes global->LDS (2 x B128); ASYNCcnt-tracked, no VGPR staging.
__device__ __forceinline__ void async_cp32(unsigned lds, const _Float16* g) {
  asm volatile("global_load_async_to_lds_b128 %0, %1, off"
               :: "v"(lds), "v"(g) : "memory");
  asm volatile("global_load_async_to_lds_b128 %0, %1, off offset:16"
               :: "v"(lds), "v"(g) : "memory");
}
__device__ __forceinline__ void wait_async0() {
  asm volatile("s_wait_asynccnt 0x0" ::: "memory");
}
__device__ __forceinline__ unsigned lds_u32(const void* p) {
  return (unsigned)(uintptr_t)p;   // low 32 bits of flat LDS address = wave-relative offset
}

// ---------------------------------------------------------------- convert (straight)
__global__ __launch_bounds__(256) void cvt_f32_f16(const float* __restrict__ in,
                                                   _Float16* __restrict__ out, int n4) {
  int i = blockIdx.x * 256 + threadIdx.x;
  if (i < n4) {
    f32x4 x = ((const f32x4*)in)[i];
    out[i*4+0] = (_Float16)x.x;
    out[i*4+1] = (_Float16)x.y;
    out[i*4+2] = (_Float16)x.z;
    out[i*4+3] = (_Float16)x.w;
  }
}

// ---------------------------------------------------------------- transpose-convert
// in: [K][N] f32  ->  out: [N][K] f16   (32x32 LDS tiles, coalesced both sides)
__global__ __launch_bounds__(256) void transpose_f32_f16(
    const float* __restrict__ in, _Float16* __restrict__ out, int K, int N) {
  __shared__ float tile[32][33];
  const int k0 = blockIdx.y * 32, n0 = blockIdx.x * 32;
  const int tx = threadIdx.x & 31, ty = threadIdx.x >> 5;   // 32 x 8
#pragma unroll
  for (int r = 0; r < 32; r += 8)
    tile[ty + r][tx] = in[(size_t)(k0 + ty + r) * N + n0 + tx];
  __syncthreads();
#pragma unroll
  for (int r = 0; r < 32; r += 8)
    out[(size_t)(n0 + ty + r) * K + k0 + tx] = (_Float16)tile[tx][ty + r];
}

// ---------------------------------------------------------------- GEMM (WMMA f16)
// C[M,N] = A[M,K] * Bt[N,K]^T ; both operands K-contiguous.
// 256 threads = 8 waves; block tile 128x128, K-chunk 32 (double-buffered async);
// wave tile 32x64.
#define GAP 40    // LDS pitch (halves) for both tiles

__global__ __launch_bounds__(256) void gemm_wmma_f16(
    const _Float16* __restrict__ A,   // [M,K] f16 row-major
    const _Float16* __restrict__ Bt,  // [N,K] f16 row-major (pre-transposed weights)
    const float*    __restrict__ bias,// [N] f32
    int M, int N, int K, int mode, float qscale,
    _Float16* __restrict__ qb, _Float16* __restrict__ kb, _Float16* __restrict__ vb,
    float* __restrict__ outp)
{
  __shared__ _Float16 As[2][128 * GAP];
  __shared__ _Float16 Bs[2][128 * GAP];

  const int tid  = threadIdx.x;
  const int wid  = tid >> 5;
  const int lane = tid & 31;
  const int hi   = lane >> 4;
  const int ln   = lane & 15;
  const int m0 = blockIdx.y * 128, n0 = blockIdx.x * 128;
  const int wm = (wid >> 1) * 32,  wn = (wid & 1) * 64;

  v8f vzero = {};
  v8f acc[2][4];
#pragma unroll
  for (int i = 0; i < 2; i++)
#pragma unroll
    for (int j = 0; j < 4; j++) acc[i][j] = vzero;

  const int row = tid >> 1, seg = tid & 1;   // 128 rows x 2 segs of 16 halves

  const _Float16* gA = A  + (size_t)(m0 + row) * K + seg * 16;
  const _Float16* gB = Bt + (size_t)(n0 + row) * K + seg * 16;
  unsigned ldsA[2], ldsB[2];
#pragma unroll
  for (int p = 0; p < 2; p++) {
    ldsA[p] = lds_u32(&As[p][row * GAP + seg * 16]);
    ldsB[p] = lds_u32(&Bs[p][row * GAP + seg * 16]);
  }

  const int nk = K >> 5;
  // prologue: chunk 0 -> buffer 0
  async_cp32(ldsA[0], gA);
  async_cp32(ldsB[0], gB);
  wait_async0();
  __syncthreads();

  for (int i = 0; i < nk; i++) {
    const int cur = i & 1, nxt = cur ^ 1;
    if (i + 1 < nk) {                         // overlap next chunk's DMA with compute
      async_cp32(ldsA[nxt], gA + (i + 1) * 32);
      async_cp32(ldsB[nxt], gB + (i + 1) * 32);
      if (i + 2 < nk) {
        __builtin_prefetch(gA + (i + 2) * 32, 0, 3);
        __builtin_prefetch(gB + (i + 2) * 32, 0, 3);
      }
    }

    Frag16 af[2], bf[4];
#pragma unroll
    for (int ms = 0; ms < 2; ms++) {           // A-fragment half pattern
      const _Float16* p = &As[cur][(wm + ms * 16 + ln) * GAP];
      af[ms].q[0] = *(const u32x4*)(p + hi * 8);
      af[ms].q[1] = *(const u32x4*)(p + 16 + hi * 8);
    }
#pragma unroll
    for (int ns = 0; ns < 4; ns++) {           // B-fragment half pattern
      const _Float16* p = &Bs[cur][(wn + ns * 16 + ln) * GAP];
      bf[ns].q[0] = *(const u32x4*)(p + hi * 8);
      bf[ns].q[1] = *(const u32x4*)(p + 16 + hi * 8);
    }
#pragma unroll
    for (int ms = 0; ms < 2; ms++)
#pragma unroll
      for (int ns = 0; ns < 4; ns++)
        acc[ms][ns] = __builtin_amdgcn_wmma_f32_16x16x32_f16(
            false, af[ms].v, false, bf[ns].v, (short)0, acc[ms][ns], false, false);

    wait_async0();       // own async writes (next chunk) landed in LDS
    __syncthreads();     // everyone's landed; also fences reads of 'cur'
  }

  // ---- epilogue
  float bn[4];
#pragma unroll
  for (int ns = 0; ns < 4; ns++) bn[ns] = bias[n0 + wn + ns * 16 + ln];

#pragma unroll
  for (int ms = 0; ms < 2; ms++) {
#pragma unroll
    for (int ns = 0; ns < 4; ns++) {
      const int gn = n0 + wn + ns * 16 + ln;
#pragma unroll
      for (int j = 0; j < 8; j++) {
        const int gm = m0 + wm + ms * 16 + j + hi * 8;
        float val = acc[ms][ns][j] + bn[ns];
        if (mode == 0) {
          int which = gn >> 10;        // / C
          int c = gn & (Cc - 1);
          int h = c >> 6, d = c & (Dd - 1);
          int b = gm >> 11;            // / T
          int t = gm & (Tt - 1);
          if (which == 0) val *= qscale;
          if (which == 2) {            // V stored transposed: [B,H,D,T]
            vb[(((size_t)(b * Hh + h)) * Dd + d) * Tt + t] = (_Float16)val;
          } else {
            _Float16* dstb = (which == 0) ? qb : kb;
            dstb[(((size_t)(b * Hh + h)) * Tt + t) * Dd + d] = (_Float16)val;
          }
        } else {
          outp[(size_t)gm * N + gn] = val;
        }
      }
    }
  }
}

// ---------------------------------------------------------------- Flash attention
// 128 threads = 4 waves. Each wave: 16 query rows, full D=64. Key blocks of 32,
// double-buffered async K/V tiles.
// qbuf,kbuf: [B,H,T,D] f16 ; vbuf: [B,H,D,T] f16 (pre-transposed)
#define KP 72   // K LDS pitch (halves)
#define VP 40   // V^T LDS pitch (halves)
#define PP 40   // P LDS pitch (halves)

__global__ __launch_bounds__(128) void attn_wmma(
    const _Float16* __restrict__ qbuf, const _Float16* __restrict__ kbuf,
    const _Float16* __restrict__ vbuf, _Float16* __restrict__ ybuf)
{
  __shared__ _Float16 Kt[2][32 * KP];     // [key][d]
  __shared__ _Float16 Vs[2][64 * VP];     // [d][key]  (transposed tile)
  __shared__ _Float16 Pl[4 * 16 * PP];    // per-wave [q][key]

  const int tid  = threadIdx.x;
  const int w    = tid >> 5;
  const int lane = tid & 31;
  const int hi   = lane >> 4;
  const int ln   = lane & 15;
  const int qt = blockIdx.x, bh = blockIdx.y;
  const int b = bh >> 4, h = bh & (Hh - 1);
  const int q0 = qt * 64;

  // Q fragments (A-operand half pattern), d-chunks of 32
  Frag16 qa[2];
  {
    const int tq = q0 + w * 16 + ln;
    const _Float16* qp = qbuf + ((size_t)bh * Tt + tq) * Dd;
#pragma unroll
    for (int c = 0; c < 2; c++) {
      qa[c].q[0] = *(const u32x4*)(qp + c * 32 + hi * 8);
      qa[c].q[1] = *(const u32x4*)(qp + c * 32 + 16 + hi * 8);
    }
  }

  float mrow[8], lrow[8];
  v8f vzero = {};
  v8f ya[4];
#pragma unroll
  for (int j = 0; j < 8; j++) { mrow[j] = -1e30f; lrow[j] = 0.0f; }
#pragma unroll
  for (int c = 0; c < 4; c++) ya[c] = vzero;

  const int krow = tid >> 2, kseg = tid & 3;   // K tile: 32 rows x 4 segs of 16 halves
  const int vrow = tid >> 1, vseg = tid & 1;   // V^T tile: 64 rows x 2 segs
  const int nkb = qt * 2 + 2;
  _Float16* Pw = Pl + w * 16 * PP;

  const _Float16* gK = kbuf + ((size_t)bh * Tt + krow) * Dd + kseg * 16;
  const _Float16* gV = vbuf + ((size_t)bh * Dd + vrow) * Tt + vseg * 16;
  unsigned ldsK[2], ldsV[2];
#pragma unroll
  for (int p = 0; p < 2; p++) {
    ldsK[p] = lds_u32(&Kt[p][krow * KP + kseg * 16]);
    ldsV[p] = lds_u32(&Vs[p][vrow * VP + vseg * 16]);
  }

  // prologue: key block 0 -> buffer 0
  async_cp32(ldsK[0], gK);
  async_cp32(ldsV[0], gV);
  wait_async0();
  __syncthreads();

  for (int kb = 0; kb < nkb; kb++) {
    const int cur = kb & 1, nxt = cur ^ 1;
    if (kb + 1 < nkb) {
      async_cp32(ldsK[nxt], gK + (size_t)(kb + 1) * 32 * Dd);  // next 32 key rows
      async_cp32(ldsV[nxt], gV + (kb + 1) * 32);               // next 32 key cols
    }

    // wave-uniform skip of fully-masked key blocks
    if (kb * 32 <= q0 + w * 16 + 15) {
      // S = Q K^T for two 16-key sub-blocks
      v8f s[2];
#pragma unroll
      for (int sub = 0; sub < 2; sub++) {
        v8f sc = vzero;
#pragma unroll
        for (int dc = 0; dc < 2; dc++) {
          Frag16 kf;                                   // B-operand: half pattern
          const _Float16* kp = &Kt[cur][(sub * 16 + ln) * KP + dc * 32];
          kf.q[0] = *(const u32x4*)(kp + hi * 8);
          kf.q[1] = *(const u32x4*)(kp + 16 + hi * 8);
          sc = __builtin_amdgcn_wmma_f32_16x16x32_f16(
              false, qa[dc].v, false, kf.v, (short)0, sc, false, false);
        }
        s[sub] = sc;
      }

      // causal mask + online softmax
      const int qrow_base = q0 + w * 16 + hi * 8;
#pragma unroll
      for (int j = 0; j < 8; j++) {
        int qr = qrow_base + j;
        bool msk0 = (kb * 32 + ln) > qr;
        bool msk1 = (kb * 32 + 16 + ln) > qr;
        float s0 = msk0 ? -1e30f : s[0][j];
        float s1 = msk1 ? -1e30f : s[1][j];
        float rm = fmaxf(s0, s1);
        rm = fmaxf(rm, __shfl_xor(rm, 1));
        rm = fmaxf(rm, __shfl_xor(rm, 2));
        rm = fmaxf(rm, __shfl_xor(rm, 4));
        rm = fmaxf(rm, __shfl_xor(rm, 8));
        float mnew  = fmaxf(mrow[j], rm);
        float alpha = __expf(mrow[j] - mnew);
        mrow[j] = mnew;
        float p0 = msk0 ? 0.0f : __expf(s0 - mnew);
        float p1 = msk1 ? 0.0f : __expf(s1 - mnew);
        float rs = p0 + p1;
        rs += __shfl_xor(rs, 1);
        rs += __shfl_xor(rs, 2);
        rs += __shfl_xor(rs, 4);
        rs += __shfl_xor(rs, 8);
        lrow[j] = lrow[j] * alpha + rs;
#pragma unroll
        for (int c = 0; c < 4; c++) ya[c][j] *= alpha;
        Pw[(j + hi * 8) * PP + ln]      = (_Float16)p0;
        Pw[(j + hi * 8) * PP + 16 + ln] = (_Float16)p1;
      }
      asm volatile("s_wait_dscnt 0x0" ::: "memory");

      // reload P as A-fragment
      Frag16 pf;
      {
        const _Float16* pp = Pw + ln * PP;
        pf.q[0] = *(const u32x4*)(pp + hi * 8);
        pf.q[1] = *(const u32x4*)(pp + 16 + hi * 8);
      }
      // Y += P x V : B-operand from transposed V tile, half pattern
#pragma unroll
      for (int c = 0; c < 4; c++) {
        Frag16 vf;
        const _Float16* vp = &Vs[cur][(c * 16 + ln) * VP];
        vf.q[0] = *(const u32x4*)(vp + hi * 8);
        vf.q[1] = *(const u32x4*)(vp + 16 + hi * 8);
        ya[c] = __builtin_amdgcn_wmma_f32_16x16x32_f16(
            false, pf.v, false, vf.v, (short)0, ya[c], false, false);
      }
    }

    wait_async0();       // next K/V tile landed (own lanes)
    __syncthreads();     // all waves done writing + done reading 'cur'
  }

  // normalize + store y into [B,T,C] f16
#pragma unroll
  for (int j = 0; j < 8; j++) {
    float inv = 1.0f / lrow[j];
    int tq = q0 + w * 16 + hi * 8 + j;
    size_t base = ((size_t)b * Tt + tq) * Cc + h * Dd;
#pragma unroll
    for (int c = 0; c < 4; c++)
      ybuf[base + c * 16 + ln] = (_Float16)(ya[c][j] * inv);
  }
}

// ---------------------------------------------------------------- launcher
extern "C" void kernel_launch(void* const* d_in, const int* in_sizes, int n_in,
                              void* d_out, int out_size, void* d_ws, size_t ws_size,
                              hipStream_t stream) {
  const float* x      = (const float*)d_in[0];
  const float* w_attn = (const float*)d_in[1];
  const float* b_attn = (const float*)d_in[2];
  const float* w_proj = (const float*)d_in[3];
  const float* b_proj = (const float*)d_in[4];
  float* out = (float*)d_out;

  char* ws = (char*)d_ws;
  size_t off = 0;
  auto take = [&](size_t nelem) -> _Float16* {
    _Float16* p = (_Float16*)(ws + off);
    off += ((nelem * sizeof(_Float16) + 255) & ~(size_t)255);
    return p;
  };
  _Float16* xf  = take((size_t)Mm * Cc);            // x            [M,K]
  _Float16* waT = take((size_t)3 * Cc * Cc);        // w_attn^T     [3C,C]
  _Float16* wpT = take((size_t)Cc * Cc);            // w_proj^T     [C,C]
  _Float16* qb  = take((size_t)Bb * Hh * Tt * Dd);  // q            [B,H,T,D]
  _Float16* kb  = take((size_t)Bb * Hh * Tt * Dd);  // k            [B,H,T,D]
  _Float16* vb  = take((size_t)Bb * Hh * Tt * Dd);  // v^T          [B,H,D,T]
  _Float16* yb  = take((size_t)Mm * Cc);            // attn out     [M,C]

  {
    int n4 = Mm * Cc / 4;
    cvt_f32_f16<<<(n4 + 255) / 256, 256, 0, stream>>>(x, xf, n4);
  }
  transpose_f32_f16<<<dim3(3 * Cc / 32, Cc / 32), 256, 0, stream>>>(w_attn, waT, Cc, 3 * Cc);
  transpose_f32_f16<<<dim3(Cc / 32, Cc / 32), 256, 0, stream>>>(w_proj, wpT, Cc, Cc);

  // QKV: fused bias + head-split (+ V transpose) + q-scale epilogue
  gemm_wmma_f16<<<dim3(3 * Cc / 128, Mm / 128), 256, 0, stream>>>(
      xf, waT, b_attn, Mm, 3 * Cc, Cc, /*mode=*/0, /*qscale=*/1.0f / 32.0f,
      qb, kb, vb, nullptr);

  // Flash attention over heads
  attn_wmma<<<dim3(Tt / 64, Bb * Hh), 128, 0, stream>>>(qb, kb, vb, yb);

  // Proj: + bias -> f32 out
  gemm_wmma_f16<<<dim3(Cc / 128, Mm / 128), 256, 0, stream>>>(
      yb, wpT, b_proj, Mm, Cc, Cc, /*mode=*/1, 1.0f,
      nullptr, nullptr, nullptr, out);
}
